// LMU_33586644255224
// MI455X (gfx1250) — matrix operand order repaired and checked by reading, not compiled, not run on hardware
//
#include <hip/hip_runtime.h>
#include <hip/hip_bf16.h>

// ---------------- problem constants ----------------
#define B_SZ   32
#define T_LEN  1024
#define IN_D   256
#define MEM_D  8
#define ORD    256
#define HID    512
#define NROW   (B_SZ * MEM_D)     // 256 independent recurrence rows (b*8+d)
#define CHUNK  32
#define NCHUNK (T_LEN / CHUNK)    // 32

// ---------------- vector types ----------------
typedef __attribute__((ext_vector_type(2)))  float  v2f;
typedef __attribute__((ext_vector_type(8)))  float  v8f;
typedef __attribute__((ext_vector_type(8)))  __bf16 v8bf;
typedef __attribute__((ext_vector_type(16))) __bf16 v16bf;
typedef int vi4 __attribute__((vector_size(16)));   // type the async builtin wants

// ---------------- WMMA wrappers ----------------
__device__ __forceinline__ v8f wmma_f32_k4(v2f a, v2f b, v8f c) {
  // D(16x16,f32) = A(16x4,f32) x B(4x16,f32) + C
  return __builtin_amdgcn_wmma_f32_16x16x4_f32(false, a, false, b, (short)0, c,
                                               false, false);
}
__device__ __forceinline__ v8f wmma_bf16_k32(v16bf a, v16bf b, v8f c) {
  // D(16x16,f32) = A(16x32,bf16) x B(32x16,bf16) + C
  return __builtin_amdgcn_wmma_f32_16x16x32_bf16(false, a, false, b, (short)0, c,
                                                 false, false);
}

// ---------------- async global->LDS staging (probed) ----------------
#if __has_builtin(__builtin_amdgcn_global_load_async_to_lds_b128)
#define USE_ASYNC 1
#else
#define USE_ASYNC 0
#endif

__device__ __forceinline__ void copy16_g2l(const __bf16* g, __bf16* l) {
#if USE_ASYNC
  __builtin_amdgcn_global_load_async_to_lds_b128((vi4*)g, (vi4*)l, 0, 0);
#else
  *(v8bf*)l = *(const v8bf*)g;   // global_load_b128 + ds_store_b128
#endif
}
__device__ __forceinline__ void wait_stage() {
#if USE_ASYNC
#if __has_builtin(__builtin_amdgcn_s_wait_asynccnt)
  __builtin_amdgcn_s_wait_asynccnt(0);
#else
  asm volatile("s_wait_asynccnt 0" ::: "memory");
#endif
#endif
}

// =====================================================================
// Kernel 1: U = x @ kernel, stored row-major as U[row=b*8+d][t]
// =====================================================================
__global__ __launch_bounds__(256) void lmu_u_kernel(
    const float* __restrict__ x, const float* __restrict__ kw,
    float* __restrict__ U) {
  __shared__ float Ks[IN_D * MEM_D];   // 8 KB
  for (int i = threadIdx.x; i < IN_D * MEM_D; i += 256) Ks[i] = kw[i];
  __syncthreads();
  const int bt = blockIdx.x * 256 + threadIdx.x;  // (b,t) flat, 0..32767
  const float* xr = x + (size_t)bt * IN_D;
  float acc[MEM_D];
#pragma unroll
  for (int d = 0; d < MEM_D; ++d) acc[d] = 0.0f;
  for (int i = 0; i < IN_D; ++i) {
    const float xv = xr[i];
#pragma unroll
    for (int d = 0; d < MEM_D; ++d) acc[d] += xv * Ks[i * MEM_D + d];
  }
  const int b = bt >> 10, t = bt & 1023;
#pragma unroll
  for (int d = 0; d < MEM_D; ++d)
    U[(size_t)(b * MEM_D + d) * T_LEN + t] = acc[d];
}

// =====================================================================
// Kernel 2: Wht[n][k] = bf16(Wh[k][n])  (pre-transpose so GEMM staging
// of the B operand is contiguous along K)
// =====================================================================
__global__ __launch_bounds__(256) void wh_convert_kernel(
    const float* __restrict__ Wh, __bf16* __restrict__ WhT) {
  const int idx = blockIdx.x * 256 + threadIdx.x;  // < 2048*512
  const int k = idx >> 9, n = idx & 511;
  WhT[(size_t)n * (MEM_D * ORD) + k] = (__bf16)Wh[idx];
}

// =====================================================================
// Kernel 3: D = X @ X for 256x256 fp32 (used to square A up to A^32).
// Grid: 16 WGs (one 16-row M tile each), 8 waves x 2 N tiles, K=256.
// =====================================================================
__global__ __launch_bounds__(256) void gemm256_sq_kernel(
    const float* __restrict__ X, float* __restrict__ D) {
  const int lane = threadIdx.x & 31, wave = threadIdx.x >> 5;
  const int hi = lane >> 4, lo = lane & 15;
  const int m0 = blockIdx.x * 16;
  v2f afr[64];
#pragma unroll
  for (int kt = 0; kt < 64; ++kt) {
    const float* p = X + (size_t)(m0 + lo) * ORD + kt * 4 + 2 * hi;
    afr[kt].x = p[0]; afr[kt].y = p[1];
  }
  for (int nn = 0; nn < 2; ++nn) {
    const int n0 = (wave * 2 + nn) * 16;
    v8f acc = {0.f, 0.f, 0.f, 0.f, 0.f, 0.f, 0.f, 0.f};
#pragma unroll
    for (int kt = 0; kt < 64; ++kt) {
      const int kb = kt * 4 + 2 * hi;
      const float* bp = X + (size_t)kb * ORD + n0 + lo;
      v2f bfr; bfr.x = bp[0]; bfr.y = bp[ORD];
      acc = wmma_f32_k4(afr[kt], bfr, acc);
    }
    float* q = D + (size_t)(m0 + 8 * hi) * ORD + n0 + lo;
#pragma unroll
    for (int v = 0; v < 8; ++v) q[(size_t)v * ORD] = acc[v];
  }
}

// =====================================================================
// Kernel 4 (mode 0/1): chunked recurrence.  Each wave owns 32 rows of
// state (rows are independent: NO barriers).  State slab lives in LDS;
// per step the wave hoists its whole 16x256 K-panel into 128 VGPRs of
// f32 A-fragments, then does 16 N-tiles x 64 K-tiles of
// v_wmma_f32_16x16x4_f32 with the rank-1 u(x)Brow update folded into
// the accumulator init.
//   mode 0: zero init, write end state E[chunk]
//   mode 1: init from Sin[chunk] (carry-in), write every m_t to ms (bf16)
// Grid: 32 chunks x 4 row-groups = 128 WGs of 64 threads (2 waves).
// =====================================================================
__global__ __launch_bounds__(64) void lmu_chunk_kernel(
    const float* __restrict__ U,     // [NROW][T_LEN]
    const float* __restrict__ Amat,  // [ORD][ORD] (o-major: m@A)
    const float* __restrict__ Brow,  // [ORD]
    const float* __restrict__ Sin,   // [NCHUNK][NROW][ORD] or null
    float*       __restrict__ Eout,  // [NCHUNK][NROW][ORD] or null
    __bf16*      __restrict__ ms,    // [B*T][MEM_D*ORD] or null
    int mode) {
  __shared__ float S[64 * ORD];      // 64 rows x 256 f32 = 64 KB
  const int lane = threadIdx.x & 31, wave = threadIdx.x >> 5;
  const int hi = lane >> 4, lo = lane & 15;
  const int chunk = blockIdx.x >> 2;
  const int rg = blockIdx.x & 3;
  const int R0 = rg * 64 + wave * 32;      // first global row of this wave
  float* Sw = S + wave * 32 * ORD;

  if (mode == 0) {
    for (int i = lane; i < 32 * ORD; i += 32) Sw[i] = 0.0f;
  } else {
    const float* src = Sin + ((size_t)chunk * NROW + R0) * ORD;
    for (int i = lane; i < 32 * ORD; i += 32) Sw[i] = src[i];
  }

  const int t0 = chunk * CHUNK;
  for (int tt = 0; tt < CHUNK; ++tt) {
    const int t = t0 + tt;
    if (t + 1 < T_LEN) __builtin_prefetch(&U[(size_t)R0 * T_LEN + t + 1], 0, 0);
    for (int mt = 0; mt < 2; ++mt) {
      const int m0w = wave * 32 + mt * 16;   // LDS row base
      const int m0g = R0 + mt * 16;          // global row base
      // per-lane u values for its 8 accumulator row slots
      float uv[8];
      const int rbase = m0g + 8 * hi;
#pragma unroll
      for (int v = 0; v < 8; ++v) uv[v] = U[(size_t)(rbase + v) * T_LEN + t];
      // hoist the full K panel of this M tile (safe to overwrite after)
      v2f afr[64];
#pragma unroll
      for (int kt = 0; kt < 64; ++kt) {
        const float* p = S + (m0w + lo) * ORD + kt * 4 + 2 * hi;
        afr[kt].x = p[0]; afr[kt].y = p[1];
      }
      for (int nt = 0; nt < 16; ++nt) {
        const int n0 = nt * 16;
        const float brw = Brow[n0 + lo];
        v8f acc;
#pragma unroll
        for (int v = 0; v < 8; ++v) acc[v] = uv[v] * brw;  // C = u (x) Brow
#pragma unroll
        for (int kt = 0; kt < 64; ++kt) {
          const int kb = kt * 4 + 2 * hi;
          const float* bp = Amat + (size_t)kb * ORD + n0 + lo;
          v2f bfr; bfr.x = bp[0]; bfr.y = bp[ORD];
          acc = wmma_f32_k4(afr[kt], bfr, acc);
        }
        float* q = S + (m0w + 8 * hi) * ORD + n0 + lo;
#pragma unroll
        for (int v = 0; v < 8; ++v) q[(size_t)v * ORD] = acc[v];
        if (mode == 1) {
#pragma unroll
          for (int v = 0; v < 8; ++v) {
            const int r = m0g + 8 * hi + v;            // r = b*8+d
            const int col = (r & 7) * ORD + n0 + lo;   // d*ORD + p
            ms[((size_t)(r >> 3) * T_LEN + t) * (MEM_D * ORD) + col] =
                (__bf16)acc[v];
          }
        }
      }
    }
  }
  if (mode == 0) {
    float* dst = Eout + ((size_t)chunk * NROW + R0) * ORD;
    for (int i = lane; i < 32 * ORD; i += 32) dst[i] = Sw[i];
  }
}

// =====================================================================
// Kernel 5: carry scan across chunks.  Sout[k] = carry-in for chunk k:
//   S_0 = 0 ; S_{k+1} = S_k @ A^32 + E_k   (32 short sequential steps)
// Rows independent -> 4 WGs x 2 waves, no barriers.
// =====================================================================
__global__ __launch_bounds__(64) void lmu_carry_kernel(
    const float* __restrict__ Apow,  // A^CHUNK [ORD][ORD]
    const float* __restrict__ E,     // [NCHUNK][NROW][ORD]
    float*       __restrict__ Sout)  // [NCHUNK][NROW][ORD]
{
  __shared__ float S[64 * ORD];      // 64 KB
  const int lane = threadIdx.x & 31, wave = threadIdx.x >> 5;
  const int hi = lane >> 4, lo = lane & 15;
  const int R0 = blockIdx.x * 64 + wave * 32;
  float* Sw = S + wave * 32 * ORD;
  for (int i = lane; i < 32 * ORD; i += 32) Sw[i] = 0.0f;

  for (int k = 0; k < NCHUNK; ++k) {
    float* dst = Sout + ((size_t)k * NROW + R0) * ORD;
    for (int i = lane; i < 32 * ORD; i += 32) dst[i] = Sw[i];
    for (int mt = 0; mt < 2; ++mt) {
      const int m0w = wave * 32 + mt * 16;
      const int m0g = R0 + mt * 16;
      v2f afr[64];
#pragma unroll
      for (int kt = 0; kt < 64; ++kt) {
        const float* p = S + (m0w + lo) * ORD + kt * 4 + 2 * hi;
        afr[kt].x = p[0]; afr[kt].y = p[1];
      }
      for (int nt = 0; nt < 16; ++nt) {
        const int n0 = nt * 16;
        const float* ep =
            E + ((size_t)k * NROW + m0g + 8 * hi) * ORD + n0 + lo;
        v8f acc;
#pragma unroll
        for (int v = 0; v < 8; ++v) acc[v] = ep[(size_t)v * ORD];
#pragma unroll
        for (int kt = 0; kt < 64; ++kt) {
          const int kb = kt * 4 + 2 * hi;
          const float* bp = Apow + (size_t)kb * ORD + n0 + lo;
          v2f bfr; bfr.x = bp[0]; bfr.y = bp[ORD];
          acc = wmma_f32_k4(afr[kt], bfr, acc);
        }
        float* q = S + (m0w + 8 * hi) * ORD + n0 + lo;
#pragma unroll
        for (int v = 0; v < 8; ++v) q[(size_t)v * ORD] = acc[v];
      }
    }
  }
}

// =====================================================================
// Kernel 6: out = tanh(ms @ Wh + bh), bf16 WMMA, M=32768 N=512 K=2048.
// 128x128 block tile, BK=64, double-buffered LDS (64 KB) with async
// global->LDS staging; 8 waves as 4(M)x2(N), each wave 32x64.
// =====================================================================
__global__ __launch_bounds__(256) void lmu_out_gemm_kernel(
    const __bf16* __restrict__ msA,  // [32768][2048]
    const __bf16* __restrict__ WhT,  // [512][2048]
    const float* __restrict__ bh,    // [512]
    float* __restrict__ out) {       // [32768][512]
  __shared__ __bf16 Asm[2][128 * 64];   // 2 x 16 KB
  __shared__ __bf16 Bsm[2][128 * 64];   // 2 x 16 KB
  const int tid = threadIdx.x, lane = tid & 31, wave = tid >> 5;
  const int hi = lane >> 4, lo = lane & 15;
  const int bm = blockIdx.x >> 2, bn = blockIdx.x & 3;
  const int wm = (wave & 3) * 32, wn = (wave >> 2) * 64;
  const __bf16* gA = msA + (size_t)bm * 128 * 2048;
  const __bf16* gB = WhT + (size_t)bn * 128 * 2048;

  const v8f zero8 = {0.f, 0.f, 0.f, 0.f, 0.f, 0.f, 0.f, 0.f};
  v8f acc[2][4];
#pragma unroll
  for (int mt = 0; mt < 2; ++mt)
#pragma unroll
    for (int nt = 0; nt < 4; ++nt) acc[mt][nt] = zero8;

  const int srow = tid >> 3;            // staging row 0..31 (+i*32)
  const int sch = (tid & 7) * 8;        // staging 8-half chunk within row

#pragma unroll
  for (int i = 0; i < 4; ++i) {         // prologue: stage kb=0
    const int row = srow + i * 32;
    copy16_g2l(gA + (size_t)row * 2048 + sch, &Asm[0][row * 64 + sch]);
    copy16_g2l(gB + (size_t)row * 2048 + sch, &Bsm[0][row * 64 + sch]);
  }

  for (int kb = 0; kb < 32; ++kb) {
    wait_stage();
    __syncthreads();
    const int cur = kb & 1;
    if (kb + 1 < 32) {
      const int nxt = cur ^ 1;
      const size_t koff = (size_t)(kb + 1) * 64;
#pragma unroll
      for (int i = 0; i < 4; ++i) {
        const int row = srow + i * 32;
        copy16_g2l(gA + (size_t)row * 2048 + koff + sch,
                   &Asm[nxt][row * 64 + sch]);
        copy16_g2l(gB + (size_t)row * 2048 + koff + sch,
                   &Bsm[nxt][row * 64 + sch]);
      }
    }
    const __bf16* As = Asm[cur];
    const __bf16* Bs = Bsm[cur];
#pragma unroll
    for (int ks = 0; ks < 2; ++ks) {
      v16bf af[2], bf[4];
#pragma unroll
      for (int mt = 0; mt < 2; ++mt) {   // A-frag: K groups +0..8 and +16..24
        const __bf16* p = As + (wm + mt * 16 + lo) * 64 + ks * 32 + hi * 8;
        union { v16bf v; v8bf h[2]; } u;
        u.h[0] = *(const v8bf*)p;
        u.h[1] = *(const v8bf*)(p + 16);
        af[mt] = u.v;
      }
#pragma unroll
      for (int nt = 0; nt < 4; ++nt) {   // B-frag: K groups +0..8 and +16..24
        const __bf16* p = Bs + (wn + nt * 16 + lo) * 64 + ks * 32 + hi * 8;
        union { v16bf v; v8bf h[2]; } u;
        u.h[0] = *(const v8bf*)p;
        u.h[1] = *(const v8bf*)(p + 16);
        bf[nt] = u.v;
      }
#pragma unroll
      for (int mt = 0; mt < 2; ++mt)
#pragma unroll
        for (int nt = 0; nt < 4; ++nt)
          acc[mt][nt] = wmma_bf16_k32(af[mt], bf[nt], acc[mt][nt]);
    }
  }
#pragma unroll
  for (int mt = 0; mt < 2; ++mt)
#pragma unroll
    for (int nt = 0; nt < 4; ++nt)
#pragma unroll
      for (int v = 0; v < 8; ++v) {
        const int row = bm * 128 + wm + mt * 16 + 8 * hi + v;
        const int col = bn * 128 + wn + nt * 16 + lo;
        out[(size_t)row * HID + col] = tanhf(acc[mt][nt][v] + bh[col]);
      }
}

// =====================================================================
// Host launcher
// =====================================================================
extern "C" void kernel_launch(void* const* d_in, const int* in_sizes, int n_in,
                              void* d_out, int out_size, void* d_ws,
                              size_t ws_size, hipStream_t stream) {
  (void)in_sizes; (void)n_in; (void)out_size;
  const float* x    = (const float*)d_in[0];  // [32][1024][256]
  const float* kw   = (const float*)d_in[1];  // [256][8]
  const float* Wh   = (const float*)d_in[2];  // [2048][512]
  const float* bh   = (const float*)d_in[3];  // [512]
  const float* Amat = (const float*)d_in[4];  // [256][256]
  const float* Brow = (const float*)d_in[5];  // [256]
  float* out = (float*)d_out;

  char* w = (char*)d_ws;
  size_t off = 0;
  auto take = [&](size_t bytes) {
    size_t o = off;
    off += (bytes + 255) & ~(size_t)255;
    return o;
  };
  const size_t oU  = take((size_t)NROW * T_LEN * 4);            //   1 MB
  const size_t oA0 = take((size_t)ORD * ORD * 4);               // 256 KB
  const size_t oA1 = take((size_t)ORD * ORD * 4);               // 256 KB
  const size_t oE  = take((size_t)NCHUNK * NROW * ORD * 4);     //   8 MB
  const size_t oS  = take((size_t)NCHUNK * NROW * ORD * 4);     //   8 MB
  const size_t oWt = take((size_t)HID * MEM_D * ORD * 2);       //   2 MB
  const size_t oMs = take((size_t)B_SZ * T_LEN * MEM_D * ORD * 2); // 128 MB
  if (ws_size < off) return;  // workspace too small: bail deterministically

  float*  Urows = (float*)(w + oU);
  float*  A0    = (float*)(w + oA0);
  float*  A1    = (float*)(w + oA1);
  float*  Eb    = (float*)(w + oE);
  float*  Sb    = (float*)(w + oS);
  __bf16* Wt    = (__bf16*)(w + oWt);
  __bf16* Ms    = (__bf16*)(w + oMs);

  wh_convert_kernel<<<4096, 256, 0, stream>>>(Wh, Wt);
  lmu_u_kernel<<<128, 256, 0, stream>>>(x, kw, Urows);
  // A^2, A^4, A^8, A^16, A^32 by repeated squaring (result in A0)
  gemm256_sq_kernel<<<16, 256, 0, stream>>>(Amat, A0);
  gemm256_sq_kernel<<<16, 256, 0, stream>>>(A0, A1);
  gemm256_sq_kernel<<<16, 256, 0, stream>>>(A1, A0);
  gemm256_sq_kernel<<<16, 256, 0, stream>>>(A0, A1);
  gemm256_sq_kernel<<<16, 256, 0, stream>>>(A1, A0);
  // phase 1: per-chunk end states from zero init
  lmu_chunk_kernel<<<128, 64, 0, stream>>>(Urows, Amat, Brow, nullptr, Eb,
                                           nullptr, 0);
  // phase 2: carry scan -> carry-in state per chunk
  lmu_carry_kernel<<<4, 64, 0, stream>>>(A0, Eb, Sb);
  // phase 3: replay chunks from carry-in, emit ms (bf16)
  lmu_chunk_kernel<<<128, 64, 0, stream>>>(Urows, Amat, Brow, Sb, nullptr, Ms,
                                           1);
  // phase 4: out = tanh(ms @ Wh + bh)
  lmu_out_gemm_kernel<<<1024, 256, 0, stream>>>(Ms, Wt, bh, out);
}
